// LearnedStructuredAttention_10814727651437
// MI455X (gfx1250) — matrix operand order repaired and checked by reading, not compiled, not run on hardware
//
#include <hip/hip_runtime.h>

#define SEQ 2048
#define BATCH 4
#define HID 1024

typedef __attribute__((ext_vector_type(16))) __bf16 v16bf;
typedef __attribute__((ext_vector_type(8)))  float  v8f;
typedef __attribute__((ext_vector_type(4)))  unsigned int v4u;

// round-to-nearest-even fp32 -> bf16 bits
__device__ __forceinline__ unsigned short f2bf(float f) {
  unsigned int u = __float_as_uint(f);
  u += 0x7FFFu + ((u >> 16) & 1u);
  return (unsigned short)(u >> 16);
}

// 32-byte async copy global -> LDS (two b128, ASYNCcnt += 2)
__device__ __forceinline__ void async_copy32(unsigned lds_addr, const unsigned short* g) {
  asm volatile("global_load_async_to_lds_b128 %0, %1, off\n\t"
               "global_load_async_to_lds_b128 %0, %1, off offset:16"
               :: "v"(lds_addr), "v"(g) : "memory");
}

// Block GEMM: 128(M) x 128(N) tile, K consumed in 32-wide double-buffered stages
// staged into LDS with async copies. 8 waves: wave w owns rows [m0blk+16w, +16).
// A is [M][K] row-major; Bt is [N][K] row-major (B columns = Bt rows).
// acc[t] accumulates the 16x16 tile at columns n0blk + 16t  (t = 0..7).
__device__ __forceinline__ void gemm_block_lds(const unsigned short* __restrict__ A, int lda,
                                               const unsigned short* __restrict__ Bt, int ldb,
                                               int m0blk, int n0blk, int K, v8f acc[8]) {
  constexpr int AP = 40;                    // padded LDS row pitch (80B): 16B-aligned, conflict-free
  constexpr int TILE_ELEMS = 128 * AP;      // one 128x32 tile (padded)
  constexpr int BUF_ELEMS = 2 * TILE_ELEMS; // A tile + B tile
  __shared__ unsigned short smem[2 * BUF_ELEMS];  // double buffered: 40 KB

  const int tid = threadIdx.x;
  const int lane = tid & 31, w = tid >> 5;
  const unsigned base = (unsigned)(unsigned long long)(const void*)smem;  // LDS byte address
  const unsigned bufBytes = BUF_ELEMS * 2;

  // staging: each tile is 128 rows x 32 elems; thread -> (row = tid/2, 32B half = tid&1)
  const int row = tid >> 1, half = tid & 1;
  const unsigned short* aG = A  + (size_t)(m0blk + row) * lda + half * 16;
  const unsigned short* bG = Bt + (size_t)(n0blk + row) * ldb + half * 16;
  const unsigned aL = base + (unsigned)(row * AP + half * 16) * 2;
  const unsigned bL = aL + (unsigned)TILE_ELEMS * 2;

  auto stage = [&](int kt, int dd) {
    const unsigned off = (unsigned)dd * bufBytes;
    const int ke = kt * 32;
    async_copy32(aL + off, aG + ke);   // 2 async ops
    async_copy32(bL + off, bG + ke);   // 2 async ops
  };

  const int fm = lane & 15, fg = lane >> 4;
  auto compute = [&](int dd) {
    const unsigned short* aBuf = smem + dd * BUF_ELEMS;
    const unsigned short* bBuf = aBuf + TILE_ELEMS;
    // A fragment 16x32: lane m = L&15, g = L>>4; halves 0..7 = K g*8+j, 8..15 = K 16+g*8+j
    const unsigned short* pa = aBuf + (w * 16 + fm) * AP + fg * 8;
    union { v4u u[2]; v16bf v; } ca;
    ca.u[0] = *(const v4u*)pa;
    ca.u[1] = *(const v4u*)(pa + 16);
    // load ALL B fragments into distinct registers before any WMMA issues
    union { v4u u[2]; v16bf v; } cb[8];
#pragma unroll
    for (int t = 0; t < 8; ++t) {
      // B fragment 32x16: lane n = L&15, g = L>>4; halves 0..7 = K g*8+j, 8..15 = K 16+g*8+j of column n
      const unsigned short* pb = bBuf + (t * 16 + fm) * AP + fg * 8;
      cb[t].u[0] = *(const v4u*)pb;
      cb[t].u[1] = *(const v4u*)(pb + 16);
    }
    // hard scheduling fence: loads may not be sunk past this point, so the WMMAs
    // below wait only on progressively older ds_loads instead of dscnt==0 each.
    __builtin_amdgcn_sched_barrier(0);
#pragma unroll
    for (int t = 0; t < 8; ++t)
      acc[t] = __builtin_amdgcn_wmma_f32_16x16x32_bf16(false, ca.v, false, cb[t].v,
                                                       (short)0, acc[t], false, false);
  };

  const int nK = K / 32;
  stage(0, 0);
  int d = 0;
  // steady state is branch-free; tail stage peeled below
#pragma clang loop unroll(disable)
  for (int kt = 0; kt < nK - 1; ++kt) {
    stage(kt + 1, d ^ 1);
    // allow next stage's 4 copies in flight; wait for current stage (in-order completion)
    asm volatile("s_wait_asynccnt 0x4" ::: "memory");
    __syncthreads();
    compute(d);
    __syncthreads();   // compiler emits s_wait_dscnt 0 before the barrier -> safe to overwrite
    d ^= 1;
  }
  asm volatile("s_wait_asynccnt 0x0" ::: "memory");
  __syncthreads();
  compute(d);
}

__global__ void k_convert(const float* __restrict__ src, unsigned short* __restrict__ dst, int n) {
  int i = blockIdx.x * blockDim.x + threadIdx.x;
  int stride = gridDim.x * blockDim.x;
  for (; i < n; i += stride) dst[i] = f2bf(src[i]);
}

// QKV projection: y[m][n] = sum_k Xb[m][k] * W[n][k] + bias[n]
// m = s*BATCH + b.  which = blockIdx.z: 0->Qb[b][s][h], 1->Kb[b][s][h], 2->Vt[b][h][s]
__global__ void __launch_bounds__(256, 1)
k_qkv(const unsigned short* __restrict__ Xb,
      const unsigned short* __restrict__ Wb,
      const float* __restrict__ bq, const float* __restrict__ bk,
      const float* __restrict__ bv,
      unsigned short* __restrict__ Qb, unsigned short* __restrict__ Kb,
      unsigned short* __restrict__ Vt) {
  const int lane = threadIdx.x & 31, w = threadIdx.x >> 5;
  const int m0blk = blockIdx.x * 128;
  const int n0 = blockIdx.y * 128;
  const int which = blockIdx.z;
  const unsigned short* W = Wb + (size_t)which * HID * HID;
  const float* bias = (which == 0) ? bq : ((which == 1) ? bk : bv);

  v8f acc[8] = {};
  gemm_block_lds(Xb, HID, W, HID, m0blk, n0, HID, acc);

  const int m0 = m0blk + w * 16;
  const int nl = lane & 15, rb = (lane >> 4) * 8;
#pragma unroll
  for (int t = 0; t < 8; ++t)
#pragma unroll
    for (int r = 0; r < 8; ++r) {
      int m = m0 + rb + r;
      int col = n0 + 16 * t + nl;
      float val = acc[t][r] + bias[col];
      int s = m >> 2, b = m & 3;
      unsigned short bv16 = f2bf(val);
      if (which == 0)      Qb[((size_t)b * SEQ + s) * HID + col] = bv16;
      else if (which == 1) Kb[((size_t)b * SEQ + s) * HID + col] = bv16;
      else                 Vt[((size_t)b * HID + col) * SEQ + s] = bv16;
    }
}

// scores[b][q][kk] = (Q[b][q] . K[b][kk]) / 32 * ssm[b][q][kk]
__global__ void __launch_bounds__(256, 1)
k_scores(const unsigned short* __restrict__ Qb,
         const unsigned short* __restrict__ Kb,
         const float* __restrict__ ssm,
         float* __restrict__ Sc) {
  const int lane = threadIdx.x & 31, w = threadIdx.x >> 5;
  const int b = blockIdx.z;
  const int m0blk = blockIdx.x * 128;
  const int n0 = blockIdx.y * 128;
  const unsigned short* A  = Qb + (size_t)b * SEQ * HID;
  const unsigned short* Bt = Kb + (size_t)b * SEQ * HID;

  v8f acc[8] = {};
  gemm_block_lds(A, HID, Bt, HID, m0blk, n0, HID, acc);

  const float scale = 0.03125f;  // 1/sqrt(1024)
  const int m0 = m0blk + w * 16;
  const int nl = lane & 15, rb = (lane >> 4) * 8;
#pragma unroll
  for (int t = 0; t < 8; ++t)
#pragma unroll
    for (int r = 0; r < 8; ++r) {
      int q = m0 + rb + r;
      int kk = n0 + 16 * t + nl;
      size_t idx = ((size_t)b * SEQ + q) * SEQ + kk;
      Sc[idx] = acc[t][r] * scale * ssm[idx];
    }
}

// row-wise softmax over 2048, write bf16 probabilities
__global__ void __launch_bounds__(256)
k_softmax(const float* __restrict__ Sc, unsigned short* __restrict__ P) {
  __shared__ float red[256];
  int q = blockIdx.x, b = blockIdx.y;
  int t = threadIdx.x;
  const float* row = Sc + ((size_t)b * SEQ + q) * SEQ;

  float v[8];
  float m = -1e30f;
#pragma unroll
  for (int i = 0; i < 8; ++i) {
    v[i] = row[t + i * 256];
    m = fmaxf(m, v[i]);
  }
  red[t] = m; __syncthreads();
  for (int s = 128; s > 0; s >>= 1) {
    if (t < s) red[t] = fmaxf(red[t], red[t + s]);
    __syncthreads();
  }
  m = red[0]; __syncthreads();

  float sum = 0.f;
#pragma unroll
  for (int i = 0; i < 8; ++i) {
    v[i] = __expf(v[i] - m);
    sum += v[i];
  }
  red[t] = sum; __syncthreads();
  for (int s = 128; s > 0; s >>= 1) {
    if (t < s) red[t] += red[t + s];
    __syncthreads();
  }
  float inv = 1.0f / red[0];

  unsigned short* prow = P + ((size_t)b * SEQ + q) * SEQ;
#pragma unroll
  for (int i = 0; i < 8; ++i) prow[t + i * 256] = f2bf(v[i] * inv);
}

// out[q][b][h] = sum_kk P[b][q][kk] * Vt[b][h][kk]
__global__ void __launch_bounds__(256, 1)
k_out(const unsigned short* __restrict__ P,
      const unsigned short* __restrict__ Vt,
      float* __restrict__ out) {
  const int lane = threadIdx.x & 31, w = threadIdx.x >> 5;
  const int b = blockIdx.z;
  const int m0blk = blockIdx.x * 128;
  const int n0 = blockIdx.y * 128;
  const unsigned short* A  = P  + (size_t)b * SEQ * SEQ;
  const unsigned short* Bt = Vt + (size_t)b * HID * SEQ;

  v8f acc[8] = {};
  gemm_block_lds(A, SEQ, Bt, SEQ, m0blk, n0, SEQ, acc);

  const int m0 = m0blk + w * 16;
  const int nl = lane & 15, rb = (lane >> 4) * 8;
#pragma unroll
  for (int t = 0; t < 8; ++t)
#pragma unroll
    for (int r = 0; r < 8; ++r) {
      int q = m0 + rb + r;
      int col = n0 + 16 * t + nl;
      out[(size_t)q * (BATCH * HID) + (size_t)b * HID + col] = acc[t][r];
    }
}

extern "C" void kernel_launch(void* const* d_in, const int* in_sizes, int n_in,
                              void* d_out, int out_size, void* d_ws, size_t ws_size,
                              hipStream_t stream) {
  (void)in_sizes; (void)n_in; (void)out_size; (void)ws_size;
  const float* x   = (const float*)d_in[0];
  const float* ssm = (const float*)d_in[1];
  const float* Wq  = (const float*)d_in[2];
  const float* bq  = (const float*)d_in[3];
  const float* Wk  = (const float*)d_in[4];
  const float* bk  = (const float*)d_in[5];
  const float* Wv  = (const float*)d_in[6];
  const float* bv  = (const float*)d_in[7];
  float* out = (float*)d_out;

  // workspace layout (bf16 unless noted)
  unsigned short* Xb = (unsigned short*)d_ws;                    // [8192][1024]
  unsigned short* Wb = Xb + (size_t)SEQ * BATCH * HID;           // 3 x [1024][1024]
  unsigned short* Qb = Wb + (size_t)3 * HID * HID;               // [4][2048][1024]
  unsigned short* Kb = Qb + (size_t)BATCH * SEQ * HID;           // [4][2048][1024]
  unsigned short* Vt = Kb + (size_t)BATCH * SEQ * HID;           // [4][1024][2048]
  float*          Sc = (float*)(Vt + (size_t)BATCH * SEQ * HID); // [4][2048][2048] fp32
  unsigned short* P  = (unsigned short*)(Sc + (size_t)BATCH * SEQ * SEQ); // [4][2048][2048]

  k_convert<<<512, 256, 0, stream>>>(x,  Xb, SEQ * BATCH * HID);
  k_convert<<<512, 256, 0, stream>>>(Wq, Wb,                 HID * HID);
  k_convert<<<512, 256, 0, stream>>>(Wk, Wb +     HID * HID, HID * HID);
  k_convert<<<512, 256, 0, stream>>>(Wv, Wb + 2 * HID * HID, HID * HID);

  k_qkv<<<dim3(64, 8, 3), 256, 0, stream>>>(Xb, Wb, bq, bk, bv, Qb, Kb, Vt);
  k_scores<<<dim3(16, 16, 4), 256, 0, stream>>>(Qb, Kb, ssm, Sc);
  k_softmax<<<dim3(SEQ, BATCH), 256, 0, stream>>>(Sc, P);
  k_out<<<dim3(16, 8, 4), 256, 0, stream>>>(P, Vt, out);
}
